// CausalSelfAttention_3109556322344
// MI455X (gfx1250) — matrix operand order repaired and checked by reading, not compiled, not run on hardware
//
#include <hip/hip_runtime.h>
#include <hip/hip_bf16.h>

#define B_   2
#define T_   2048
#define C_   1024
#define H_   16
#define HD_  64
#define MTOK (B_ * T_)   // 4096 tokens

typedef __bf16 bf16;
typedef __attribute__((ext_vector_type(16))) __bf16       bf16x16;
typedef __attribute__((ext_vector_type(8)))  float        f32x8;
typedef __attribute__((ext_vector_type(4)))  unsigned int u32x4;

union Frag {
    bf16x16 v;
    u32x4   q[2];
};

__device__ __forceinline__ f32x8 wmma_bf16(bf16x16 a, bf16x16 b, f32x8 c) {
    // D = A(16x32 bf16) x B(32x16 bf16) + C(16x16 f32)
    return __builtin_amdgcn_wmma_f32_16x16x32_bf16(
        /*neg_a=*/false, a, /*neg_b=*/false, b,
        /*c_mod=*/(short)0, c, /*reuse_a=*/false, /*reuse_b=*/false);
}

// ---------------------------------------------------------------------------
// fp32 -> bf16 elementwise
// ---------------------------------------------------------------------------
__global__ __launch_bounds__(256) void cvt_f32_bf16(const float* __restrict__ src,
                                                    bf16* __restrict__ dst, int n) {
    for (int i = blockIdx.x * blockDim.x + threadIdx.x; i < n;
         i += gridDim.x * blockDim.x)
        dst[i] = (bf16)src[i];
}

// ---------------------------------------------------------------------------
// transpose + convert: WT[n][k] = (bf16) W[k][n], 1024x1024, 4 matrices (z)
// ---------------------------------------------------------------------------
__global__ __launch_bounds__(256) void transpose_cvt(
    const float* __restrict__ W0, const float* __restrict__ W1,
    const float* __restrict__ W2, const float* __restrict__ W3,
    bf16* __restrict__ T0, bf16* __restrict__ T1,
    bf16* __restrict__ T2, bf16* __restrict__ T3) {
    const float* src;
    bf16* dst;
    switch (blockIdx.z) {
        case 0:  src = W0; dst = T0; break;
        case 1:  src = W1; dst = T1; break;
        case 2:  src = W2; dst = T2; break;
        default: src = W3; dst = T3; break;
    }
    __shared__ float tile[32][33];
    const int tx  = threadIdx.x & 31;
    const int ty0 = threadIdx.x >> 5;  // 0..7
    const int col = blockIdx.x * 32 + tx;
#pragma unroll
    for (int r = 0; r < 32; r += 8)
        tile[ty0 + r][tx] = src[(size_t)(blockIdx.y * 32 + ty0 + r) * C_ + col];
    __syncthreads();
    const int ocol = blockIdx.y * 32 + tx;
#pragma unroll
    for (int r = 0; r < 32; r += 8)
        dst[(size_t)(blockIdx.x * 32 + ty0 + r) * C_ + ocol] =
            (bf16)tile[tx][ty0 + r];
}

// ---------------------------------------------------------------------------
// GEMM: Y[M,N] = A[M,K](bf16) * W (given as WT[N,K] bf16) + bias, f32 accum.
//   mode 0: bf16 out, head-split [B,H,T,HD], value scaled by `scale`
//   mode 1: bf16 out, head-split transposed [B,H,HD,T]
//   mode 2: f32 out, row-major [M,N]
// One wave computes a 64x64 tile. 8 waves / block.
// ---------------------------------------------------------------------------
__global__ __launch_bounds__(256) void gemm_wmma(
    const bf16* __restrict__ A, const bf16* __restrict__ WT,
    const float* __restrict__ bias, bf16* __restrict__ dstB,
    float* __restrict__ dstF, int M, int N, int K, int mode, float scale) {
    const int lane  = threadIdx.x & 31;
    const int wave  = threadIdx.x >> 5;
    const int nT64  = N / 64;
    const int w     = blockIdx.x * 8 + wave;
    const int mBase = (w / nT64) * 64;
    const int nBase = (w % nT64) * 64;
    const int l16   = lane & 15;
    const int half  = lane >> 4;

    f32x8 acc[4][4];
#pragma unroll
    for (int mi = 0; mi < 4; mi++)
#pragma unroll
        for (int ni = 0; ni < 4; ni++)
#pragma unroll
            for (int j = 0; j < 8; j++) acc[mi][ni][j] = 0.0f;

    for (int k0 = 0; k0 < K; k0 += 32) {
        Frag a[4];
#pragma unroll
        for (int mi = 0; mi < 4; mi++) {
            // A-frag (16x32): lane half0 K=0..7,16..23 ; half1 K=8..15,24..31
            const bf16* p = A + (size_t)(mBase + mi * 16 + l16) * K + k0 + half * 8;
            a[mi].q[0] = *(const u32x4*)p;
            a[mi].q[1] = *(const u32x4*)(p + 16);
        }
#pragma unroll
        for (int ni = 0; ni < 4; ni++) {
            // B-frag (32x16): lane half0 K=0..7,16..23 ; half1 K=8..15,24..31 (from WT rows)
            Frag b;
            const bf16* p = WT + (size_t)(nBase + ni * 16 + l16) * K + k0 + half * 8;
            b.q[0] = *(const u32x4*)p;
            b.q[1] = *(const u32x4*)(p + 16);
#pragma unroll
            for (int mi = 0; mi < 4; mi++)
                acc[mi][ni] = wmma_bf16(a[mi].v, b.v, acc[mi][ni]);
        }
    }

#pragma unroll
    for (int mi = 0; mi < 4; mi++)
#pragma unroll
        for (int ni = 0; ni < 4; ni++) {
            const int n   = nBase + ni * 16 + l16;
            const float bv = bias[n];
#pragma unroll
            for (int j = 0; j < 8; j++) {
                const int m  = mBase + mi * 16 + j + half * 8;  // D layout: M = j + 8*half
                const float v = (acc[mi][ni][j] + bv) * scale;
                if (mode == 2) {
                    dstF[(size_t)m * N + n] = v;
                } else {
                    const int bidx = m / T_, t = m % T_;
                    const int h = n / HD_, d = n % HD_;
                    if (mode == 0)
                        dstB[(((size_t)bidx * H_ + h) * T_ + t) * HD_ + d] = (bf16)v;
                    else
                        dstB[(((size_t)bidx * H_ + h) * HD_ + d) * T_ + t] = (bf16)v;
                }
            }
        }
}

// ---------------------------------------------------------------------------
// Flash attention: one wave per (b,h, 16-query tile). Q pre-scaled by 1/sqrt(HD).
//   q,k: [B,H,T,HD] bf16    vt: [B,H,HD,T] bf16    y: [B,T,C] bf16 (heads merged)
// ---------------------------------------------------------------------------
#define AW 8
#define PSTRIDE 40  // bf16 elems per P-tile row (32 + 8 pad), 80B, 16B aligned

__global__ __launch_bounds__(256) void attn_flash(
    const bf16* __restrict__ q, const bf16* __restrict__ k,
    const bf16* __restrict__ vt, bf16* __restrict__ y) {
    __shared__ __align__(16) bf16 plds[AW][16 * PSTRIDE];
    const int lane = threadIdx.x & 31;
    const int wave = threadIdx.x >> 5;
    const int w    = blockIdx.x * AW + wave;
    const int QT   = T_ / 16;           // 128 query tiles per (b,h)
    const int bh   = w / QT;            // 0..B*H-1
    const int qt   = w % QT;
    const int b    = bh / H_;
    const int h    = bh % H_;
    const int l16  = lane & 15;
    const int half = lane >> 4;
    const int qBase = qt * 16;
    const size_t baseQK = (size_t)bh * T_ * HD_;
    const size_t baseV  = (size_t)bh * HD_ * T_;
    bf16* myP = plds[wave];

    // Q A-frags for d = 0..31 and 32..63
    Frag qf[2];
#pragma unroll
    for (int ds = 0; ds < 2; ds++) {
        const bf16* p = q + baseQK + (size_t)(qBase + l16) * HD_ + ds * 32 + half * 8;
        qf[ds].q[0] = *(const u32x4*)p;
        qf[ds].q[1] = *(const u32x4*)(p + 16);
    }

    float mrow[8], lrow[8];
    f32x8 oacc[4];
#pragma unroll
    for (int j = 0; j < 8; j++) { mrow[j] = -1e30f; lrow[j] = 0.0f; }
#pragma unroll
    for (int ni = 0; ni < 4; ni++)
#pragma unroll
        for (int j = 0; j < 8; j++) oacc[ni][j] = 0.0f;

    const int nsteps = (qBase + 16 + 31) / 32;  // causal: keys < qBase+16
    for (int kt = 0; kt < nsteps; kt++) {
        const int kB = kt * 32;
        // --- S = Q * K^T (two 16-key subtiles, K-dim = HD in 2 steps) ---
        f32x8 s[2];
#pragma unroll
        for (int tk = 0; tk < 2; tk++) {
            Frag bk0, bk1;
            const bf16* p0 = k + baseQK + (size_t)(kB + tk * 16 + l16) * HD_ + 0 * 32 + half * 8;
            const bf16* p1 = k + baseQK + (size_t)(kB + tk * 16 + l16) * HD_ + 1 * 32 + half * 8;
            bk0.q[0] = *(const u32x4*)p0;  bk0.q[1] = *(const u32x4*)(p0 + 16);
            bk1.q[0] = *(const u32x4*)p1;  bk1.q[1] = *(const u32x4*)(p1 + 16);
            f32x8 z;
#pragma unroll
            for (int j = 0; j < 8; j++) z[j] = 0.0f;
            z = wmma_bf16(qf[0].v, bk0.v, z);
            z = wmma_bf16(qf[1].v, bk1.v, z);
            s[tk] = z;
        }
        // --- causal mask, online softmax row stats (rows live in 16-lane groups) ---
        float alpha[8];
#pragma unroll
        for (int j = 0; j < 8; j++) {
            const int qrow = qBase + j + half * 8;
            if (kB + l16 > qrow)      s[0][j] = -1e30f;
            if (kB + 16 + l16 > qrow) s[1][j] = -1e30f;
            float t = fmaxf(s[0][j], s[1][j]);
            t = fmaxf(t, __shfl_xor(t, 1));
            t = fmaxf(t, __shfl_xor(t, 2));
            t = fmaxf(t, __shfl_xor(t, 4));
            t = fmaxf(t, __shfl_xor(t, 8));
            const float mnew = fmaxf(mrow[j], t);
            alpha[j] = __expf(mrow[j] - mnew);
            const float p0 = __expf(s[0][j] - mnew);
            const float p1 = __expf(s[1][j] - mnew);
            float rs = p0 + p1;
            rs += __shfl_xor(rs, 1);
            rs += __shfl_xor(rs, 2);
            rs += __shfl_xor(rs, 4);
            rs += __shfl_xor(rs, 8);
            lrow[j] = lrow[j] * alpha[j] + rs;
            mrow[j] = mnew;
            // stage P (bf16) in LDS: D-layout -> memory tile [16][32]
            const int r = j + half * 8;
            myP[r * PSTRIDE + l16]      = (bf16)p0;
            myP[r * PSTRIDE + 16 + l16] = (bf16)p1;
        }
        // rescale O accumulators
#pragma unroll
        for (int ni = 0; ni < 4; ni++)
#pragma unroll
            for (int j = 0; j < 8; j++) oacc[ni][j] *= alpha[j];

        asm volatile("s_wait_dscnt 0" ::: "memory");

        // reload P as A-frag (16x32 bf16)
        Frag pf;
        {
            const bf16* p = myP + l16 * PSTRIDE + half * 8;
            pf.q[0] = *(const u32x4*)p;
            pf.q[1] = *(const u32x4*)(p + 16);
        }
        // --- O += P * V  (V transposed: contiguous keys per d-row) ---
#pragma unroll
        for (int ni = 0; ni < 4; ni++) {
            Frag bv;
            const bf16* p = vt + baseV + (size_t)(ni * 16 + l16) * T_ + kB + half * 8;
            bv.q[0] = *(const u32x4*)p;
            bv.q[1] = *(const u32x4*)(p + 16);
            oacc[ni] = wmma_bf16(pf.v, bv.v, oacc[ni]);
        }
    }

    // epilogue: divide by l, merge heads into [B,T,C]
#pragma unroll
    for (int j = 0; j < 8; j++) {
        const float inv = 1.0f / lrow[j];
        const int t = qBase + j + half * 8;
#pragma unroll
        for (int ni = 0; ni < 4; ni++)
            y[((size_t)b * T_ + t) * C_ + h * HD_ + ni * 16 + l16] =
                (bf16)(oacc[ni][j] * inv);
    }
}

// ---------------------------------------------------------------------------
extern "C" void kernel_launch(void* const* d_in, const int* in_sizes, int n_in,
                              void* d_out, int out_size, void* d_ws, size_t ws_size,
                              hipStream_t stream) {
    const float* x  = (const float*)d_in[0];
    const float* Wq = (const float*)d_in[1];
    const float* bq = (const float*)d_in[2];
    const float* Wk = (const float*)d_in[3];
    const float* bk = (const float*)d_in[4];
    const float* Wv = (const float*)d_in[5];
    const float* bv = (const float*)d_in[6];
    const float* Wp = (const float*)d_in[7];
    const float* bp = (const float*)d_in[8];
    float* out = (float*)d_out;

    char* ws = (char*)d_ws;
    const size_t xBytes = (size_t)MTOK * C_ * sizeof(bf16);  // 8 MB
    const size_t wBytes = (size_t)C_ * C_ * sizeof(bf16);    // 2 MB
    bf16* xb   = (bf16*)ws;               ws += xBytes;
    bf16* WqT  = (bf16*)ws;               ws += wBytes;
    bf16* WkT  = (bf16*)ws;               ws += wBytes;
    bf16* WvT  = (bf16*)ws;               ws += wBytes;
    bf16* WpT  = (bf16*)ws;               ws += wBytes;
    bf16* qB   = (bf16*)ws;               ws += xBytes;  // [B,H,T,HD]
    bf16* kB   = (bf16*)ws;               ws += xBytes;  // [B,H,T,HD]
    bf16* vtB  = (bf16*)ws;               ws += xBytes;  // [B,H,HD,T]
    bf16* yatt = (bf16*)ws;               ws += xBytes;  // [B,T,C]

    cvt_f32_bf16<<<1024, 256, 0, stream>>>(x, xb, MTOK * C_);
    transpose_cvt<<<dim3(C_ / 32, C_ / 32, 4), 256, 0, stream>>>(
        Wq, Wk, Wv, Wp, WqT, WkT, WvT, WpT);

    const int gemmBlocks = (MTOK / 64) * (C_ / 64) / 8;  // 128
    const float qscale = 0.125f;                         // 1/sqrt(HD)
    gemm_wmma<<<gemmBlocks, 256, 0, stream>>>(xb, WqT, bq, qB,  nullptr, MTOK, C_, C_, 0, qscale);
    gemm_wmma<<<gemmBlocks, 256, 0, stream>>>(xb, WkT, bk, kB,  nullptr, MTOK, C_, C_, 0, 1.0f);
    gemm_wmma<<<gemmBlocks, 256, 0, stream>>>(xb, WvT, bv, vtB, nullptr, MTOK, C_, C_, 1, 1.0f);

    attn_flash<<<(B_ * H_ * (T_ / 16)) / AW, 256, 0, stream>>>(qB, kB, vtB, yatt);

    gemm_wmma<<<gemmBlocks, 256, 0, stream>>>(yatt, WpT, bp, nullptr, out, MTOK, C_, C_, 2, 1.0f);
}
